// IsolationForestNN_12137577579169
// MI455X (gfx1250) — matrix operand order repaired and checked by reading, not compiled, not run on hardware
//
#include <hip/hip_runtime.h>
#include <hip/hip_bf16.h>
#include <math.h>

// ---------------------------------------------------------------------------
// IsolationForestNN on MI455X (gfx1250): bf16 WMMA, f32 accumulate.
//  - pack kernel: W1/W2 -> bf16, transposed for WMMA B-layout (once/launch)
//  - main kernel: grid (128 row-blocks x 4 tree-groups), 256 thr (8 waves);
//    weights async-copied global->LDS, 20 v_wmma per wave per tree.
//  - reduce kernel: deterministic sum of 4 tree-group partials.
// ---------------------------------------------------------------------------

typedef __bf16 v16bf __attribute__((ext_vector_type(16)));
typedef float  v8f   __attribute__((ext_vector_type(8)));
typedef int    v4i   __attribute__((ext_vector_type(4)));

#define NROWS 16384
#define DIM   128
#define NH1   64
#define NH2   32
#define NTREE 100
#define TGRP  4      // tree groups (grid.y)
#define TPG   25     // trees per group
#define RT    128    // rows per block
#define SX    136    // LDS stride (bf16 elems) for Xs / W1t : 272 B rows
#define S1    72     // LDS stride (bf16 elems) for H1s / W2t: 144 B rows
#define S2    33     // LDS stride (floats) for H2s

#ifndef __has_builtin
#define __has_builtin(x) 0
#endif
#if __has_builtin(__builtin_amdgcn_global_load_async_to_lds_b128) && \
    __has_builtin(__builtin_amdgcn_s_wait_asynccnt)
#define HAVE_ASYNC_LDS 1
#else
#define HAVE_ASYNC_LDS 0
#endif

// 16-byte global -> LDS copy; async (no VGPR round trip) when available.
__device__ __forceinline__ void cp16_g2l(const void* g, void* l) {
#if HAVE_ASYNC_LDS
  __builtin_amdgcn_global_load_async_to_lds_b128((v4i*)g, (v4i*)l, 0, 0);
#else
  *(uint4*)l = *(const uint4*)g;
#endif
}
__device__ __forceinline__ void cp_wait() {
#if HAVE_ASYNC_LDS
  __builtin_amdgcn_s_wait_asynccnt(0);
#endif
}

union FragU { uint4 q[2]; v16bf v; };

// A fragment (16x32 bf16): lane m=l&15 holds row m;
// elems 0..7 = K k0+8*kh+0..7, elems 8..15 = K k0+16+8*kh+0..7
__device__ __forceinline__ v16bf load_frag_a(const __bf16* rowp, int k0, int kh) {
  FragU u;
  u.q[0] = *(const uint4*)(rowp + k0 + kh * 8);
  u.q[1] = *(const uint4*)(rowp + k0 + 16 + kh * 8);
  return u.v;
}

// B fragment (32x16 bf16): lane n=l&15 holds column n;
// elems 0..7 = K k0+8*kh+0..7, elems 8..15 = K k0+16+8*kh+0..7
__device__ __forceinline__ v16bf load_frag_b(const __bf16* colp, int k0, int kh) {
  FragU u;
  u.q[0] = *(const uint4*)(colp + k0 + kh * 8);
  u.q[1] = *(const uint4*)(colp + k0 + 16 + kh * 8);
  return u.v;
}

// --------------------------- weight pack kernel ----------------------------
// w1pk[t][h1][d] = bf16(W1[t][d][h1]) ; w2pk[t][h2][h1] = bf16(W2[t][h1][h2])
__global__ __launch_bounds__(256) void iforest_pack(
    const float* __restrict__ W1, const float* __restrict__ W2,
    __bf16* __restrict__ w1pk, __bf16* __restrict__ w2pk)
{
  const int t = blockIdx.x;
  const float* W1p = W1 + (size_t)t * DIM * NH1;
  __bf16*      o1  = w1pk + (size_t)t * NH1 * DIM;
  for (int i = threadIdx.x; i < NH1 * DIM; i += 256) {
    const int h = i >> 7, d = i & (DIM - 1);
    o1[i] = (__bf16)W1p[d * NH1 + h];
  }
  const float* W2p = W2 + (size_t)t * NH1 * NH2;
  __bf16*      o2  = w2pk + (size_t)t * NH2 * NH1;
  for (int i = threadIdx.x; i < NH2 * NH1; i += 256) {
    const int h2 = i >> 6, h1 = i & (NH1 - 1);
    o2[i] = (__bf16)W2p[h1 * NH2 + h2];
  }
}

// ------------------------------ main kernel --------------------------------
__global__ __launch_bounds__(256) void iforest_kernel(
    const float* __restrict__ x,
    const __bf16* __restrict__ w1pk, const __bf16* __restrict__ w2pk,
    const float* __restrict__ b1, const float* __restrict__ b2,
    const float* __restrict__ W3, const float* __restrict__ b3,
    float* __restrict__ partial)
{
  __shared__ __align__(16) __bf16 Xs [RT  * SX];  // x tile, bf16
  __shared__ __align__(16) __bf16 W1t[NH1 * SX];  // [h1][d]
  __shared__ __align__(16) __bf16 H1s[RT  * S1];  // relu(x@W1+b1)
  __shared__ __align__(16) __bf16 W2t[NH2 * S1];  // [h2][h1]
  __shared__ __align__(16) float  H2s[RT  * S2];  // relu(h1@W2+b2)
  __shared__ float b1s[NH1];
  __shared__ float b2s[NH2];
  __shared__ float W3s[NH2];
  __shared__ float b3s;

  const int tid = threadIdx.x;
  const int n0  = blockIdx.x * RT;

  // Stage x tile once (fp32 -> bf16), coalesced global, conflict-free LDS.
  for (int i = tid; i < RT * DIM; i += 256) {
    const int r = i >> 7, d = i & (DIM - 1);
    Xs[r * SX + d] = (__bf16)x[(size_t)(n0 + r) * DIM + d];
  }

  const float cn = 2.0f * (logf((float)(NROWS - 1)) + 0.5772f)
                 - 2.0f * (float)(NROWS - 1) / (float)NROWS;
  const float inv_cn = 1.0f / cn;

  const int lane    = tid & 31;
  const int wave    = tid >> 5;
  const int m16     = lane & 15;
  const int kh      = lane >> 4;
  const int rowbase = wave * 16;

  float acc = 0.0f;
  const int t0 = blockIdx.y * TPG;

  for (int jt = 0; jt < TPG; ++jt) {
    const int t = t0 + jt;
    __syncthreads();  // protect previous iteration's LDS readers

    // Bulk-copy prepacked bf16 weights into padded LDS (16 B chunks).
    const __bf16* w1g = w1pk + (size_t)t * NH1 * DIM;
    #pragma unroll
    for (int it = 0; it < (NH1 * DIM / 8) / 256; ++it) {   // 1024 chunks
      const int c = it * 256 + tid;
      const int row = c >> 4, off = (c & 15) * 8;
      cp16_g2l(w1g + row * DIM + off, &W1t[row * SX + off]);
    }
    {
      const __bf16* w2g = w2pk + (size_t)t * NH2 * NH1;    // 256 chunks
      const int row = tid >> 3, off = (tid & 7) * 8;
      cp16_g2l(w2g + row * NH1 + off, &W2t[row * S1 + off]);
    }
    if (tid < NH1) b1s[tid] = b1[t * NH1 + tid];
    if (tid < NH2) { b2s[tid] = b2[t * NH2 + tid]; W3s[tid] = W3[t * NH2 + tid]; }
    if (tid == 0)  b3s = b3[t];
    cp_wait();
    __syncthreads();

    // ---------------- layer 1: [16x128] @ [128x64] per wave ----------------
    v8f c1[4] = {};
    #pragma unroll
    for (int kc = 0; kc < 4; ++kc) {
      const v16bf a = load_frag_a(&Xs[(rowbase + m16) * SX], kc * 32, kh);
      #pragma unroll
      for (int ct = 0; ct < 4; ++ct) {
        const v16bf bm = load_frag_b(&W1t[(ct * 16 + m16) * SX], kc * 32, kh);
        c1[ct] = __builtin_amdgcn_wmma_f32_16x16x32_bf16(
            false, a, false, bm, (short)0, c1[ct], false, false);
      }
    }
    #pragma unroll
    for (int ct = 0; ct < 4; ++ct) {
      const float bv = b1s[ct * 16 + m16];
      #pragma unroll
      for (int jj = 0; jj < 8; ++jj) {
        const float h = fmaxf(c1[ct][jj] + bv, 0.0f);
        H1s[(rowbase + jj + 8 * kh) * S1 + ct * 16 + m16] = (__bf16)h;
      }
    }
    // Wave reads back only its own 16 rows of H1s (same-wave LDS RAW).

    // ---------------- layer 2: [16x64] @ [64x32] per wave ------------------
    v8f c2[2] = {};
    #pragma unroll
    for (int kc = 0; kc < 2; ++kc) {
      const v16bf a = load_frag_a(&H1s[(rowbase + m16) * S1], kc * 32, kh);
      #pragma unroll
      for (int ct = 0; ct < 2; ++ct) {
        const v16bf bm = load_frag_b(&W2t[(ct * 16 + m16) * S1], kc * 32, kh);
        c2[ct] = __builtin_amdgcn_wmma_f32_16x16x32_bf16(
            false, a, false, bm, (short)0, c2[ct], false, false);
      }
    }
    #pragma unroll
    for (int ct = 0; ct < 2; ++ct) {
      const float bv = b2s[ct * 16 + m16];
      #pragma unroll
      for (int jj = 0; jj < 8; ++jj) {
        const float h = fmaxf(c2[ct][jj] + bv, 0.0f);
        H2s[(rowbase + jj + 8 * kh) * S2 + ct * 16 + m16] = h;
      }
    }
    __syncthreads();

    // ---------------- layer 3: sigmoid(h2 . W3 + b3) -> exp2 score ---------
    if (tid < RT) {
      float s = b3s;
      #pragma unroll
      for (int k = 0; k < NH2; ++k) s += H2s[tid * S2 + k] * W3s[k];
      const float p = 1.0f / (1.0f + expf(-s));
      acc += exp2f(-p * inv_cn);
    }
  }

  // Deterministic: each (row-block, tree-group) owns a distinct slice.
  if (tid < RT) partial[(size_t)blockIdx.y * NROWS + n0 + tid] = acc;
}

__global__ __launch_bounds__(256) void iforest_reduce(
    const float* __restrict__ partial, float* __restrict__ out)
{
  const int n = blockIdx.x * 256 + threadIdx.x;
  float s = 0.0f;
  #pragma unroll
  for (int g = 0; g < TGRP; ++g) s += partial[(size_t)g * NROWS + n];
  out[n] = s * (1.0f / (float)NTREE);
}

extern "C" void kernel_launch(void* const* d_in, const int* in_sizes, int n_in,
                              void* d_out, int out_size, void* d_ws, size_t ws_size,
                              hipStream_t stream) {
  (void)in_sizes; (void)n_in; (void)out_size; (void)ws_size;
  const float* x  = (const float*)d_in[0];
  const float* W1 = (const float*)d_in[1];
  const float* b1 = (const float*)d_in[2];
  const float* W2 = (const float*)d_in[3];
  const float* b2 = (const float*)d_in[4];
  const float* W3 = (const float*)d_in[5];
  const float* b3 = (const float*)d_in[6];
  float* out      = (float*)d_out;

  // Workspace layout: [partial: TGRP*NROWS f32][w1pk: T*64*128 bf16][w2pk: T*32*64 bf16]
  char* ws = (char*)d_ws;
  float*  partial = (float*)ws;                              // 256 KB
  __bf16* w1pk = (__bf16*)(ws + (size_t)TGRP * NROWS * 4);   // 1.6 MB
  __bf16* w2pk = w1pk + (size_t)NTREE * NH1 * DIM;           // 0.4 MB

  iforest_pack<<<NTREE, 256, 0, stream>>>(W1, W2, w1pk, w2pk);

  dim3 grid(NROWS / RT, TGRP);
  iforest_kernel<<<grid, 256, 0, stream>>>(x, w1pk, w2pk, b1, b2, W3, b3, partial);
  iforest_reduce<<<NROWS / 256, 256, 0, stream>>>(partial, out);
}
